// RNN_63797444214879
// MI455X (gfx1250) — matrix operand order repaired and checked by reading, not compiled, not run on hardware
//
#include <hip/hip_runtime.h>

// Problem constants (match reference).
#define SEQ   1024
#define BATCH 128
#define IND   64
#define HIDD  256

typedef __attribute__((ext_vector_type(16))) __bf16       v16bf;
typedef __attribute__((ext_vector_type(8)))  float        v8f;
typedef __attribute__((ext_vector_type(2)))  float        v2f;
typedef __attribute__((ext_vector_type(4)))  unsigned int u32x4;

union FragBF {
    u32x4 q[2];   // two 16B chunks (ds_load_b128 each)
    v16bf h;      // bf16 WMMA fragment
};

// ---------------------------------------------------------------------------
// Phase 1: xh[s,b,h] = x[s,b,:] . W_xh[h,:] + b_xh[h] + b_hh[h]
// Written directly into d_out[0 .. S*B*H); phase 2 overwrites it in place with
// h_t. One wave per 16x16 output tile; K=64 reduction.
// Bandwidth-bound (134 MB store); f32 WMMA keeps it exact when available.
// ---------------------------------------------------------------------------
__global__ __launch_bounds__(256) void rnn_phase1(
    const float* __restrict__ x,     // (S*B, 64)
    const float* __restrict__ W_xh,  // (256, 64)
    const float* __restrict__ b_xh,  // (256)
    const float* __restrict__ b_hh,  // (256)
    float* __restrict__ xh)          // (S*B, 256)  == d_out
{
    const int lane = threadIdx.x & 31;
    const int lhi  = lane >> 4;        // 0/1: lane half
    const int llo  = lane & 15;
    const int gw   = blockIdx.x * 8 + (threadIdx.x >> 5);
    const int m0   = (gw >> 4) * 16;   // 8192 M-tiles
    const int n0   = (gw & 15) * 16;   // 16  N-tiles

    v8f acc = {};
#if __has_builtin(__builtin_amdgcn_wmma_f32_16x16x4_f32)
    // Exact f32 WMMA. A 16x4: elem e -> K = lhi*2 + e ; B 4x16: same K, N=llo.
    const float* ap = x    + (size_t)(m0 + llo) * IND + lhi * 2;
    const float* bp = W_xh + (size_t)(n0 + llo) * IND + lhi * 2;
#pragma unroll
    for (int k = 0; k < IND; k += 4) {
        v2f a = *(const v2f*)(ap + k);
        v2f b = *(const v2f*)(bp + k);
        acc = __builtin_amdgcn_wmma_f32_16x16x4_f32(
            false, a, false, b, (short)0, acc, false, false);
    }
#else
    // bf16 fallback (codegen-confirmed builtin), K = 2 blocks of 32.
    const float* ap = x    + (size_t)(m0 + llo) * IND;
    const float* bp = W_xh + (size_t)(n0 + llo) * IND;
#pragma unroll
    for (int kb = 0; kb < 2; ++kb) {
        v16bf a, b;
#pragma unroll
        for (int e = 0; e < 8; ++e) {
            a[e]     = (__bf16)ap[kb * 32 + lhi * 8 + e];
            a[e + 8] = (__bf16)ap[kb * 32 + 16 + lhi * 8 + e];
        }
#pragma unroll
        for (int e = 0; e < 16; ++e)
            b[e] = (__bf16)bp[kb * 32 + lhi * 16 + e];
        acc = __builtin_amdgcn_wmma_f32_16x16x32_bf16(
            false, a, false, b, (short)0, acc, false, false);
    }
#endif
    // Fold both biases so the scan step is just tanh(acc_hh + xh).
    const float bias = b_xh[n0 + llo] + b_hh[n0 + llo];
    float* op = xh + (size_t)m0 * HIDD + n0 + llo;
#pragma unroll
    for (int r = 0; r < 8; ++r)                 // C/D layout: M = r + lhi*8, N = llo
        op[(size_t)(r + lhi * 8) * HIDD] = acc[r] + bias;
}

// ---------------------------------------------------------------------------
// Phase 2: sequential scan. One workgroup per 16-row batch tile -> the
// recurrence closes with only a workgroup barrier. 8 waves/WG; each wave owns
// two 16-wide hidden tiles and keeps its W_hh B-fragments resident in VGPRs
// (bf16) for all 1024 steps. h state lives in LDS as bf16.
// ---------------------------------------------------------------------------
__global__ __launch_bounds__(256) void rnn_phase2(
    const float* __restrict__ W_hh,  // (256,256) f32
    float* out)                      // (S*B + B, 256): xh -> h_seq in place, tail = h_last
{
    __shared__ __align__(16) __bf16 hsh[16 * 272];  // 16 rows, padded stride 272 halves

    const int lane = threadIdx.x & 31;
    const int wave = threadIdx.x >> 5;   // 0..7
    const int lhi  = lane >> 4;
    const int llo  = lane & 15;
    const int b0   = blockIdx.x * 16;    // batch tile

    // Load + convert W_hh slices into resident bf16 B-fragments.
    // B 32x16 layout: N = llo, K = kb*32 + lhi*8 + e (e<8), kb*32 + 16 + lhi*8 + (e-8) (e>=8).
    v16bf Bf[2][8];
#pragma unroll
    for (int j = 0; j < 2; ++j) {
        const int h0 = (wave * 2 + j) * 16;
        const float* wp = W_hh + (size_t)(h0 + llo) * HIDD + lhi * 8;
#pragma unroll
        for (int kb = 0; kb < 8; ++kb) {
            v16bf f;
#pragma unroll
            for (int e = 0; e < 16; ++e) f[e] = (__bf16)wp[kb * 32 + e + (e >= 8 ? 8 : 0)];
            Bf[j][kb] = f;
        }
    }

    // h0 = 0
    for (int i = threadIdx.x; i < 16 * 272; i += 256) hsh[i] = (__bf16)0.0f;
    __syncthreads();

    for (int t = 0; t < SEQ; ++t) {
        // Pull next timestep's xh slice (16 KB) toward L2/L0 (global_prefetch_b8).
        if (t + 1 < SEQ) {
            const float* nxt =
                out + ((size_t)(t + 1) * BATCH + b0) * HIDD + threadIdx.x * 16;
            __builtin_prefetch(nxt, 0, 1);
        }

        // A 16x32 bf16 layout: M = llo; halves at K = kb*32 + lhi*8 (+0..7) and +16.
        v8f acc0 = {}, acc1 = {};
        const __bf16* hrow = hsh + llo * 272 + lhi * 8;
#pragma unroll
        for (int kb = 0; kb < 8; ++kb) {
            FragBF a;
            a.q[0] = *(const u32x4*)(hrow + kb * 32);       // ds_load_b128
            a.q[1] = *(const u32x4*)(hrow + kb * 32 + 16);  // ds_load_b128
            acc0 = __builtin_amdgcn_wmma_f32_16x16x32_bf16(
                false, a.h, false, Bf[0][kb], (short)0, acc0, false, false);
            acc1 = __builtin_amdgcn_wmma_f32_16x16x32_bf16(
                false, a.h, false, Bf[1][kb], (short)0, acc1, false, false);
        }
        __syncthreads();  // everyone finished reading h(t-1) from LDS

#pragma unroll
        for (int j = 0; j < 2; ++j) {
            const int h0 = (wave * 2 + j) * 16;
            v8f acc = j ? acc1 : acc0;
            float* xrow = out + ((size_t)t * BATCH + b0) * HIDD + h0 + llo;
#pragma unroll
            for (int r = 0; r < 8; ++r) {
                const int m  = r + lhi * 8;                 // C/D layout row
                float pre = acc[r] + xrow[(size_t)m * HIDD];  // + xh_t (+ biases, folded)
                // tanh via hardware transcendentals: v_exp_f32 + v_rcp_f32
                float e2 = __expf(2.0f * pre);
                float th = (e2 - 1.0f) * __builtin_amdgcn_rcpf(e2 + 1.0f);
                xrow[(size_t)m * HIDD] = th;                // h_seq (in place over xh)
                hsh[m * 272 + h0 + llo] = (__bf16)th;       // bf16 state for step t+1
                if (t == SEQ - 1)                           // h_last tail
                    out[(size_t)SEQ * BATCH * HIDD +
                        (size_t)(b0 + m) * HIDD + h0 + llo] = th;
            }
        }
        __syncthreads();  // h(t) fully published before next step reads it
    }
}

// ---------------------------------------------------------------------------
extern "C" void kernel_launch(void* const* d_in, const int* in_sizes, int n_in,
                              void* d_out, int out_size, void* d_ws, size_t ws_size,
                              hipStream_t stream) {
    (void)in_sizes; (void)n_in; (void)out_size; (void)d_ws; (void)ws_size;
    const float* x    = (const float*)d_in[0];  // (S,B,64)
    const float* W_hh = (const float*)d_in[1];  // (256,256)
    const float* b_hh = (const float*)d_in[2];  // (256)
    const float* W_xh = (const float*)d_in[3];  // (256,64)
    const float* b_xh = (const float*)d_in[4];  // (256)
    float* out = (float*)d_out;                 // (S*B + B, 256)

    // Phase 1: 8192 M-tiles x 16 N-tiles, 1 wave per tile, 8 waves per block.
    rnn_phase1<<<16384, 256, 0, stream>>>(x, W_xh, b_xh, b_hh, out);
    // Phase 2: one workgroup per 16-row batch tile.
    rnn_phase2<<<8, 256, 0, stream>>>(W_hh, out);
}
